// BridgeBlock_12060268167526
// MI455X (gfx1250) — matrix operand order repaired and checked by reading, not compiled, not run on hardware
//
#include <hip/hip_runtime.h>

// ---------------------------------------------------------------------------
// MI455X (gfx1250, wave32) implementation of the bridge block.
// All GEMMs + both attention matmuls use v_wmma_f32_16x16x32_bf16
// (bf16 inputs, fp32 accumulate).  Roofline: ~250 GFLOP of GEMM vs ~100s of
// MB of traffic @ 23.3 TB/s  =>  matrix-pipe bound  =>  bf16 WMMA path.
// GEMM: 128x64 block tile, 8 waves (one WGP-filling 256-thread block),
// templated epilogue (no per-element pointer branches), next-tile
// global_prefetch_b8.  fp32->bf16 uses native casts (v_cvt_pk_bf16_f32)
// instead of bit-twiddling to keep VALU pressure off the WMMA loops.
// ---------------------------------------------------------------------------

typedef __attribute__((ext_vector_type(16))) __bf16 v16bf;
typedef __attribute__((ext_vector_type(8)))  __bf16 bf16x8;
typedef __attribute__((ext_vector_type(8)))  float  v8f;

#define D_MODEL 1024
#define N_HEADS 16
#define D_FF    4096
#define BOT     64
#define BB      4
#define TT      1024
#define M_ROWS  (BB * TT)   // 4096

static __device__ __forceinline__ __bf16 f2bf(float f) {
    return static_cast<__bf16>(f);   // native v_cvt (RNE)
}

static __device__ __forceinline__ v16bf comb8(bf16x8 lo, bf16x8 hi) {
    return __builtin_shufflevector(lo, hi, 0,1,2,3,4,5,6,7,8,9,10,11,12,13,14,15);
}

static __device__ __forceinline__ v8f wmma_bf16(v16bf a, v16bf b, v8f c) {
    return __builtin_amdgcn_wmma_f32_16x16x32_bf16(false, a, false, b, (short)0, c, false, false);
}

static __device__ __forceinline__ float gelu_erf(float x) {
    return 0.5f * x * (1.0f + erff(x * 0.70710678118654752f));
}

// ---------------------------------------------------------------------------
// LayerNorm: one 256-thread block per row of 1024.
// ---------------------------------------------------------------------------
__global__ __launch_bounds__(256)
void ln_kernel(const float* __restrict__ x, const float* __restrict__ g,
               const float* __restrict__ b, float* __restrict__ y)
{
    __shared__ float s1[256], s2[256];
    const int row = blockIdx.x, tid = threadIdx.x;
    const float4 v = *(const float4*)(x + (size_t)row * D_MODEL + tid * 4);
    s1[tid] = v.x + v.y + v.z + v.w;
    s2[tid] = v.x*v.x + v.y*v.y + v.z*v.z + v.w*v.w;
    __syncthreads();
    for (int ofs = 128; ofs > 0; ofs >>= 1) {
        if (tid < ofs) { s1[tid] += s1[tid + ofs]; s2[tid] += s2[tid + ofs]; }
        __syncthreads();
    }
    const float mean = s1[0] * (1.0f / D_MODEL);
    const float var  = s2[0] * (1.0f / D_MODEL) - mean * mean;
    const float rs   = rsqrtf(var + 1e-5f);
    const float4 gg = *(const float4*)(g + tid * 4);
    const float4 bb = *(const float4*)(b + tid * 4);
    float4 o;
    o.x = (v.x - mean) * rs * gg.x + bb.x;
    o.y = (v.y - mean) * rs * gg.y + bb.y;
    o.z = (v.z - mean) * rs * gg.z + bb.z;
    o.w = (v.w - mean) * rs * gg.w + bb.w;
    *(float4*)(y + (size_t)row * D_MODEL + tid * 4) = o;
}

// ---------------------------------------------------------------------------
// GEMM: out = epilogue(A[M,K] @ W[K,N]).  Block tile 128x64, K-step 32.
// 256 threads = 8 wave32; wave w computes rows [w*16, w*16+16) x 64 cols
// (4 WMMA accumulators).  fp32 -> bf16 conversion staged through LDS;
// fragments read back as ds_load_b128.  Epilogue compile-time specialized.
// ---------------------------------------------------------------------------
template<bool HAS_BIAS, bool HAS_RES1, bool HAS_RES2, bool GELU>
__global__ __launch_bounds__(256)
void gemm_wmma(const float* __restrict__ A, const float* __restrict__ W,
               const float* __restrict__ bias,
               const float* __restrict__ res1, const float* __restrict__ res2,
               float* __restrict__ out, int M, int N, int K)
{
    __shared__ alignas(16) __bf16 As [128][32];   // [m][k]
    __shared__ alignas(16) __bf16 Bst[ 64][32];   // [n][k] (transposed)

    const int tid  = threadIdx.x;
    const int lane = tid & 31, wave = tid >> 5;
    const int n16  = lane & 15, half = lane >> 4;
    const int m0   = blockIdx.y * 128, n0 = blockIdx.x * 64;

    v8f acc[4];
#pragma unroll
    for (int t = 0; t < 4; ++t)
#pragma unroll
        for (int r = 0; r < 8; ++r) acc[t][r] = 0.0f;

    const int nk = K >> 5;
    for (int kt = 0; kt < nk; ++kt) {
        const int k0 = kt * 32;
        // A tile: 128x32 floats = 1024 float4, coalesced along k
#pragma unroll
        for (int i = 0; i < 4; ++i) {
            const int idx4 = tid + i * 256;            // 0..1023
            const int mm = idx4 >> 3, kv = idx4 & 7;
            const float4 f = *(const float4*)(A + (size_t)(m0 + mm) * K + k0 + kv * 4);
            As[mm][kv*4+0] = f2bf(f.x); As[mm][kv*4+1] = f2bf(f.y);
            As[mm][kv*4+2] = f2bf(f.z); As[mm][kv*4+3] = f2bf(f.w);
        }
        // B tile: 32x64 floats = 512 float4, coalesced along n, stored transposed
#pragma unroll
        for (int i = 0; i < 2; ++i) {
            const int idx4 = tid + i * 256;            // 0..511
            const int kk = idx4 >> 4, nv = idx4 & 15;
            const float4 f = *(const float4*)(W + (size_t)(k0 + kk) * N + n0 + nv * 4);
            Bst[nv*4+0][kk] = f2bf(f.x); Bst[nv*4+1][kk] = f2bf(f.y);
            Bst[nv*4+2][kk] = f2bf(f.z); Bst[nv*4+3][kk] = f2bf(f.w);
        }
        // prefetch next K-tile while WMMAs run (lowers to global_prefetch_b8)
        if (kt + 1 < nk) {
            const int k1 = k0 + 32;
            __builtin_prefetch(A + (size_t)(m0 + (tid >> 1)) * K + k1 + (tid & 1) * 16, 0, 1);
            __builtin_prefetch(W + (size_t)(k1 + (tid >> 3)) * N + n0 + (tid & 7) * 8, 0, 1);
        }
        __syncthreads();

        // A fragment: lane row m = wave*16 + n16; elem i: K = (i/8)*16 + half*8 + i%8
        const int m = wave * 16 + n16;
        const v16bf af = comb8(*(const bf16x8*)&As[m][half * 8],
                               *(const bf16x8*)&As[m][16 + half * 8]);
#pragma unroll
        for (int t = 0; t < 4; ++t) {
            // B fragment: lane col n; elem i: K = (i/8)*16 + half*8 + i%8
            const int n = t * 16 + n16;
            const v16bf bf_ = comb8(*(const bf16x8*)&Bst[n][half * 8],
                                    *(const bf16x8*)&Bst[n][16 + half * 8]);
            acc[t] = wmma_bf16(af, bf_, acc[t]);
        }
        __syncthreads();
    }

    // epilogue; C/D layout: elem r -> row m0+wave*16+r+8*half, col n0+t*16+n16
#pragma unroll
    for (int t = 0; t < 4; ++t) {
#pragma unroll
        for (int r = 0; r < 8; ++r) {
            const int row = m0 + wave * 16 + r + half * 8;
            const int col = n0 + t * 16 + n16;
            float v = acc[t][r];
            if (HAS_BIAS) v += bias[col];
            if (GELU)     v  = gelu_erf(v);
            const size_t o = (size_t)row * N + col;
            if (HAS_RES1) v += res1[o];
            if (HAS_RES2) v += res2[o];
            out[o] = v;
        }
    }
}

// ---------------------------------------------------------------------------
// Flash-style attention over the packed qkv buffer [B*T, 3*D_MODEL] with
// column layout [3][H][dk].  Grid: (T/64, B*H), 128 threads = 4 waves,
// each wave owns one 16-query tile (independent; causal waves have
// different trip counts, so only per-wave LDS fences are used).
// ---------------------------------------------------------------------------
__global__ __launch_bounds__(128)
void attn_wmma(const float* __restrict__ qkv, float* __restrict__ out, int causal)
{
    __shared__ alignas(16) __bf16 Pl[4][16 * 32];   // per-wave P (16 q x 32 keys)

    const int tid  = threadIdx.x;
    const int lane = tid & 31, wave = tid >> 5;
    const int n16  = lane & 15, half = lane >> 4;
    const int bh = blockIdx.y, b = bh >> 4, h = bh & 15;
    const int q0 = blockIdx.x * 64 + wave * 16;

    // Q fragments (pre-scaled by 1/sqrt(dk)=0.125), K-dim 64 -> 2 fragments
    const size_t rowq = ((size_t)(b * TT) + q0 + n16) * (3 * D_MODEL) + h * 64;
    v16bf aq[2];
#pragma unroll
    for (int c = 0; c < 2; ++c) {
#pragma unroll
        for (int g = 0; g < 2; ++g) {
            const float4* p = (const float4*)(qkv + rowq + c * 32 + g * 16 + half * 8);
            const float4 f0 = p[0], f1 = p[1];
            aq[c][g*8+0] = f2bf(f0.x * 0.125f); aq[c][g*8+1] = f2bf(f0.y * 0.125f);
            aq[c][g*8+2] = f2bf(f0.z * 0.125f); aq[c][g*8+3] = f2bf(f0.w * 0.125f);
            aq[c][g*8+4] = f2bf(f1.x * 0.125f); aq[c][g*8+5] = f2bf(f1.y * 0.125f);
            aq[c][g*8+6] = f2bf(f1.z * 0.125f); aq[c][g*8+7] = f2bf(f1.w * 0.125f);
        }
    }

    float mrow[8], lrow[8];
    v8f acc[4];
#pragma unroll
    for (int r = 0; r < 8; ++r) { mrow[r] = -1e30f; lrow[r] = 0.0f; }
#pragma unroll
    for (int d = 0; d < 4; ++d)
#pragma unroll
        for (int r = 0; r < 8; ++r) acc[d][r] = 0.0f;

    const int nkb = causal ? ((q0 + 47) >> 5) : (TT >> 5);
    for (int kb = 0; kb < nkb; ++kb) {
        const int kbase = kb * 32;
        v8f s[2];
#pragma unroll
        for (int t = 0; t < 2; ++t)
#pragma unroll
            for (int r = 0; r < 8; ++r) s[t][r] = 0.0f;

        // scores: Q(16x64) @ K^T(64x32) -> two 16x16 tiles, 2 WMMAs each
#pragma unroll
        for (int t = 0; t < 2; ++t) {
            const size_t rk = ((size_t)(b * TT) + kbase + t * 16 + n16) * (3 * D_MODEL)
                              + D_MODEL + h * 64;
#pragma unroll
            for (int c = 0; c < 2; ++c) {
                v16bf bk;   // B frag: col = key n16; elem i: dim = c*32 + (i/8)*16 + half*8 + i%8
                const float4* p = (const float4*)(qkv + rk + c * 32 + half * 8);
                const float4 f0 = p[0], f1 = p[1], f2 = p[4], f3 = p[5];
                bk[0]=f2bf(f0.x); bk[1]=f2bf(f0.y); bk[2]=f2bf(f0.z); bk[3]=f2bf(f0.w);
                bk[4]=f2bf(f1.x); bk[5]=f2bf(f1.y); bk[6]=f2bf(f1.z); bk[7]=f2bf(f1.w);
                bk[8]=f2bf(f2.x); bk[9]=f2bf(f2.y); bk[10]=f2bf(f2.z); bk[11]=f2bf(f2.w);
                bk[12]=f2bf(f3.x); bk[13]=f2bf(f3.y); bk[14]=f2bf(f3.z); bk[15]=f2bf(f3.w);
                s[t] = wmma_bf16(aq[c], bk, s[t]);
            }
        }

        if (causal) {
#pragma unroll
            for (int t = 0; t < 2; ++t)
#pragma unroll
                for (int r = 0; r < 8; ++r) {
                    const int q   = q0 + r + half * 8;
                    const int key = kbase + t * 16 + n16;
                    s[t][r] = (key > q) ? -1e30f : s[t][r];
                }
        }

        // online softmax per row (row values live on 16 lanes of the half-group)
#pragma unroll
        for (int r = 0; r < 8; ++r) {
            float sm = fmaxf(s[0][r], s[1][r]);
            sm = fmaxf(sm, __shfl_xor(sm, 1));
            sm = fmaxf(sm, __shfl_xor(sm, 2));
            sm = fmaxf(sm, __shfl_xor(sm, 4));
            sm = fmaxf(sm, __shfl_xor(sm, 8));
            const float mn = fmaxf(mrow[r], sm);
            const float alpha = __expf(mrow[r] - mn);
            const float p0 = __expf(s[0][r] - mn);
            const float p1 = __expf(s[1][r] - mn);
            float ps = p0 + p1;
            ps += __shfl_xor(ps, 1);
            ps += __shfl_xor(ps, 2);
            ps += __shfl_xor(ps, 4);
            ps += __shfl_xor(ps, 8);
            lrow[r] = lrow[r] * alpha + ps;
            mrow[r] = mn;
#pragma unroll
            for (int d = 0; d < 4; ++d) acc[d][r] *= alpha;
            const int m = r + half * 8;
            Pl[wave][m * 32 + n16]      = f2bf(p0);
            Pl[wave][m * 32 + 16 + n16] = f2bf(p1);
        }

        // per-wave LDS fence (cross-lane P transpose; LDS ops are in-order
        // per wave, the wait also stops compiler reordering)
        asm volatile("s_wait_dscnt 0x0" ::: "memory");

        // P fragment: row m = n16; elem i: key = (i/8)*16 + half*8 + i%8
        const v16bf ap = comb8(*(const bf16x8*)&Pl[wave][n16 * 32 + half * 8],
                               *(const bf16x8*)&Pl[wave][n16 * 32 + 16 + half * 8]);
        // P(16x32) @ V(32x64) -> 4 WMMAs
#pragma unroll
        for (int d = 0; d < 4; ++d) {
            v16bf bv;   // B frag: col = dim d*16+n16; elem i: key = (i/8)*16 + half*8 + i%8
#pragma unroll
            for (int i = 0; i < 16; ++i) {
                const int key = kbase + (i >> 3) * 16 + half * 8 + (i & 7);
                bv[i] = f2bf(qkv[((size_t)(b * TT) + key) * (3 * D_MODEL)
                                 + 2 * D_MODEL + h * 64 + d * 16 + n16]);
            }
            acc[d] = wmma_bf16(ap, bv, acc[d]);
        }
        asm volatile("s_wait_dscnt 0x0" ::: "memory");
    }

    // write [B*T, D_MODEL] with col = h*64 + d
#pragma unroll
    for (int d = 0; d < 4; ++d)
#pragma unroll
        for (int r = 0; r < 8; ++r) {
            const int q   = q0 + r + half * 8;
            const int col = h * 64 + d * 16 + n16;
            out[((size_t)(b * TT) + q) * D_MODEL + col] = acc[d][r] / lrow[r];
        }
}

// ---------------------------------------------------------------------------
// Host-side orchestration.
// Input order = setup_inputs() insertion order, params dict flattened
// depth-first in insertion order:
//  0 enc_x  1 dec_x
//  2 enc_ln1_g 3 enc_ln1_b 4 enc_ln2_g 5 enc_ln2_b
//  6 w_qkv 7 w_o  8 ff_w1 9 ff_b1 10 ff_w2 11 ff_b2
//  12..17 adapt_pre_attn  {ln_g, ln_b, down_w, down_b, up_w, up_b}
//  18..23 adapt_post_attn 24..29 adapt_pre_ff 30..35 adapt_post_ff
//  36 dec_causal_mask 37 cross_mask   (masks handled analytically, unused)
// ---------------------------------------------------------------------------
struct AdapterP { const float *ln_g, *ln_b, *dw, *db, *uw, *ub; };

static void run_gemm(const float* A, const float* W, const float* bias,
                     const float* r1, const float* r2, float* out,
                     int M, int N, int K, int gelu, hipStream_t s)
{
    dim3 grid(N / 64, M / 128);
    const int key = (bias ? 8 : 0) | (r1 ? 4 : 0) | (r2 ? 2 : 0) | (gelu ? 1 : 0);
    switch (key) {
    case 0:  gemm_wmma<false,false,false,false><<<grid,256,0,s>>>(A,W,bias,r1,r2,out,M,N,K); break;
    case 4:  gemm_wmma<false,true ,false,false><<<grid,256,0,s>>>(A,W,bias,r1,r2,out,M,N,K); break;
    case 8:  gemm_wmma<true ,false,false,false><<<grid,256,0,s>>>(A,W,bias,r1,r2,out,M,N,K); break;
    case 9:  gemm_wmma<true ,false,false,true ><<<grid,256,0,s>>>(A,W,bias,r1,r2,out,M,N,K); break;
    case 12: gemm_wmma<true ,true ,false,false><<<grid,256,0,s>>>(A,W,bias,r1,r2,out,M,N,K); break;
    case 14: gemm_wmma<true ,true ,true ,false><<<grid,256,0,s>>>(A,W,bias,r1,r2,out,M,N,K); break;
    default: gemm_wmma<false,false,false,false><<<grid,256,0,s>>>(A,W,bias,r1,r2,out,M,N,K); break;
    }
}

static void run_ln(const float* x, const float* g, const float* b, float* y, hipStream_t s)
{
    ln_kernel<<<M_ROWS, 256, 0, s>>>(x, g, b, y);
}

static void run_adapter(const float* x, const AdapterP& a, const float* res2,
                        float* t_ln, float* t_bn, float* out, hipStream_t s)
{
    run_ln(x, a.ln_g, a.ln_b, t_ln, s);
    run_gemm(t_ln, a.dw, a.db, nullptr, nullptr, t_bn, M_ROWS, BOT, D_MODEL, 1, s);
    run_gemm(t_bn, a.uw, a.ub, x, res2, out, M_ROWS, D_MODEL, BOT, 0, s);
}

extern "C" void kernel_launch(void* const* d_in, const int* in_sizes, int n_in,
                              void* d_out, int out_size, void* d_ws, size_t ws_size,
                              hipStream_t stream)
{
    (void)in_sizes; (void)n_in; (void)out_size; (void)ws_size;

    const float* enc_x = (const float*)d_in[0];
    const float* dec_x = (const float*)d_in[1];
    const float* ln1_g = (const float*)d_in[2];
    const float* ln1_b = (const float*)d_in[3];
    const float* ln2_g = (const float*)d_in[4];
    const float* ln2_b = (const float*)d_in[5];
    const float* w_qkv = (const float*)d_in[6];
    const float* w_o   = (const float*)d_in[7];
    const float* ff_w1 = (const float*)d_in[8];
    const float* ff_b1 = (const float*)d_in[9];
    const float* ff_w2 = (const float*)d_in[10];
    const float* ff_b2 = (const float*)d_in[11];
    AdapterP ad[4];
    for (int i = 0; i < 4; ++i) {
        const int base = 12 + 6 * i;
        ad[i].ln_g = (const float*)d_in[base + 0];
        ad[i].ln_b = (const float*)d_in[base + 1];
        ad[i].dw   = (const float*)d_in[base + 2];
        ad[i].db   = (const float*)d_in[base + 3];
        ad[i].uw   = (const float*)d_in[base + 4];
        ad[i].ub   = (const float*)d_in[base + 5];
    }
    const AdapterP& pre_attn  = ad[0];
    const AdapterP& post_attn = ad[1];
    const AdapterP& pre_ff    = ad[2];
    const AdapterP& post_ff   = ad[3];

    float* enc_out = (float*)d_out;
    float* dec_out = enc_out + (size_t)M_ROWS * D_MODEL;

    // workspace partition (~177 MB of fp32)
    float* w = (float*)d_ws;
    float* t_ln   = w;                       w += (size_t)M_ROWS * D_MODEL;
    float* t_qkv  = w;                       w += (size_t)M_ROWS * 3 * D_MODEL;
    float* t_attn = w;                       w += (size_t)M_ROWS * D_MODEL;
    float* t_big  = w;                       w += (size_t)M_ROWS * D_FF;
    float* t_a    = w;                       w += (size_t)M_ROWS * D_MODEL;
    float* t_b    = w;                       w += (size_t)M_ROWS * D_MODEL;
    float* t_bn   = w;                       w += (size_t)M_ROWS * BOT;

    // ---------------- encoder ----------------
    run_ln(enc_x, ln1_g, ln1_b, t_ln, stream);
    run_gemm(t_ln, w_qkv, nullptr, nullptr, nullptr, t_qkv, M_ROWS, 3 * D_MODEL, D_MODEL, 0, stream);
    attn_wmma<<<dim3(TT / 64, BB * N_HEADS), 128, 0, stream>>>(t_qkv, t_attn, 0);
    run_gemm(t_attn, w_o, nullptr, enc_x, nullptr, t_a, M_ROWS, D_MODEL, D_MODEL, 0, stream); // enc1
    run_ln(t_a, ln2_g, ln2_b, t_ln, stream);
    run_gemm(t_ln, ff_w1, ff_b1, nullptr, nullptr, t_big, M_ROWS, D_FF, D_MODEL, 1, stream);
    run_gemm(t_big, ff_w2, ff_b2, t_a, nullptr, enc_out, M_ROWS, D_MODEL, D_FF, 0, stream);

    // ---------------- decoder ----------------
    // dec_adapted = adapter_pre_attn(dec_x)
    run_adapter(dec_x, pre_attn, nullptr, t_ln, t_bn, t_a, stream);
    // attn over normed dec_adapted (shared encoder weights, causal)
    run_ln(t_a, ln1_g, ln1_b, t_ln, stream);
    run_gemm(t_ln, w_qkv, nullptr, nullptr, nullptr, t_qkv, M_ROWS, 3 * D_MODEL, D_MODEL, 0, stream);
    attn_wmma<<<dim3(TT / 64, BB * N_HEADS), 128, 0, stream>>>(t_qkv, t_attn, 1);
    run_gemm(t_attn, w_o, nullptr, nullptr, nullptr, t_b, M_ROWS, D_MODEL, D_MODEL, 0, stream); // dec_attn_out
    // dec1 = dec_x + adapter_post_attn(dec_attn_out)
    run_adapter(t_b, post_attn, dec_x, t_ln, t_bn, t_a, stream);                               // t_a = dec1
    // dec_adapted2 = adapter_pre_ff(dec1)
    run_adapter(t_a, pre_ff, nullptr, t_ln, t_bn, t_b, stream);                                // t_b
    // dec_ff_out = ffn(ln2(dec_adapted2))
    run_ln(t_b, ln2_g, ln2_b, t_ln, stream);
    run_gemm(t_ln, ff_w1, ff_b1, nullptr, nullptr, t_big, M_ROWS, D_FF, D_MODEL, 1, stream);
    run_gemm(t_big, ff_w2, ff_b2, nullptr, nullptr, t_ln, M_ROWS, D_MODEL, D_FF, 0, stream);   // t_ln = dec_ff_out
    // dec_out = dec1 + adapter_post_ff(dec_ff_out)
    run_ln(t_ln, post_ff.ln_g, post_ff.ln_b, t_b, stream);
    run_gemm(t_b, post_ff.dw, post_ff.db, nullptr, nullptr, t_bn, M_ROWS, BOT, D_MODEL, 1, stream);
    run_gemm(t_bn, post_ff.uw, post_ff.ub, t_ln, t_a, dec_out, M_ROWS, D_MODEL, BOT, 0, stream);
}
